// DetectionLoss_76656576299626
// MI455X (gfx1250) — matrix operand order repaired and checked by reading, not compiled, not run on hardware
//
#include <hip/hip_runtime.h>
#include <math.h>
#include <stdint.h>

#define NCLS 80
#define A_TOT 1029
#define GTN 20
#define NO_CH 145
#define TOPKN 10
#define MASK_W 40    /* >= ceil(1029/32)=33, padded */
#define PAIR_W 2576  /* >= ceil(1029*80/32)=2573, padded */
#define TILES 65     /* ceil(1029/16) */

typedef __attribute__((ext_vector_type(16))) _Float16 v16h;
typedef __attribute__((ext_vector_type(8)))  float    v8f;
union HV16 { v16h v; _Float16 e[16]; };
union FV8  { v8f  v; float    e[8];  };

// ---------------------------------------------------------------- helpers
__device__ __forceinline__ void anchor_decode(int a, int* off, int* W, int* HW, float* st) {
  if (a < 784)      { *off = a;       *W = 28; *HW = 784; *st = 8.f;  }
  else if (a < 980) { *off = a - 784; *W = 14; *HW = 196; *st = 16.f; }
  else              { *off = a - 980; *W = 7;  *HW = 49;  *st = 32.f; }
}

__device__ __forceinline__ float ciou_f(float cx1, float cy1, float w1_, float h1_,
                                        float cx2, float cy2, float w2_, float h2_) {
  float b1x1 = cx1 - w1_ * 0.5f, b1x2 = cx1 + w1_ * 0.5f;
  float b1y1 = cy1 - h1_ * 0.5f, b1y2 = cy1 + h1_ * 0.5f;
  float b2x1 = cx2 - w2_ * 0.5f, b2x2 = cx2 + w2_ * 0.5f;
  float b2y1 = cy2 - h2_ * 0.5f, b2y2 = cy2 + h2_ * 0.5f;
  float iw = fmaxf(fminf(b1x2, b2x2) - fmaxf(b1x1, b2x1), 0.f);
  float ih = fmaxf(fminf(b1y2, b2y2) - fmaxf(b1y1, b2y1), 0.f);
  float inter = iw * ih;
  float w1 = b1x2 - b1x1, h1 = b1y2 - b1y1 + 1e-6f;
  float w2 = b2x2 - b2x1, h2 = b2y2 - b2y1 + 1e-6f;
  float uni = w1 * h1 + w2 * h2 - inter + 1e-6f;
  float iou = inter / uni;
  float cw = fmaxf(b1x2, b2x2) - fminf(b1x1, b2x1);
  float ch = fmaxf(b1y2, b2y2) - fminf(b1y1, b2y1);
  float c2 = cw * cw + ch * ch + 1e-6f;
  float dx = b2x1 + b2x2 - b1x1 - b1x2;
  float dy = b2y1 + b2y2 - b1y1 - b1y2;
  float rho2 = (dx * dx + dy * dy) * 0.25f;
  float dat = atanf(w2 / h2) - atanf(w1 / h1);
  float v = (4.f / (3.14159f * 3.14159f)) * dat * dat;
  float alpha = v / (v - iou + 1.000001f);
  return iou - (rho2 / c2 + v * alpha);
}

__device__ __forceinline__ float wave_sum(float v) {
  for (int o = 16; o; o >>= 1) v += __shfl_xor(v, o, 32);
  return v;
}

// ---------------------------------------------------------------- kernel 0: zero ws
__global__ void k_init(uint32_t* w, size_t n) {
  size_t i  = (size_t)blockIdx.x * blockDim.x + threadIdx.x;
  size_t st = (size_t)gridDim.x * blockDim.x;
  for (; i < n; i += st) w[i] = 0u;
}

// ---------------------------------------------------------------- kernel 1: assignment
// grid = B blocks of 256. Computes per-anchor nearest-gt index, and per-gt top-10
// anchors -> mask bitset + (anchor,class) pair bitset (atomicOr == set semantics).
__global__ void k_assign(const float* __restrict__ gt, const int* __restrict__ gcls,
                         uint32_t* __restrict__ gidx, uint32_t* __restrict__ maskw,
                         uint32_t* __restrict__ pairw) {
  __shared__ float cgx[GTN], cgy[GTN];
  const int b = blockIdx.x, t = threadIdx.x;
  if (t < GTN) {
    const float* gp = gt + ((size_t)b * GTN + t) * 4;
    float x = gp[0] * 224.f, y = gp[1] * 224.f, w = gp[2] * 224.f, h = gp[3] * 224.f;
    cgx[t] = x + w * 0.5f;
    cgy[t] = y + h * 0.5f;
  }
  __syncthreads();

  // per-anchor argmin over gts (first index wins on ties, matching jnp.argmin)
  for (int a = t; a < A_TOT; a += 256) {
    int off, W, HW; float st;
    anchor_decode(a, &off, &W, &HW, &st);
    float ax = ((off % W) + 0.5f) * st, ay = ((off / W) + 0.5f) * st;
    float best = 3.4e38f; int bg = 0;
    for (int g = 0; g < GTN; ++g) {
      float ddx = ax - cgx[g], ddy = ay - cgy[g];
      float d = ddx * ddx + ddy * ddy;
      if (d < best) { best = d; bg = g; }
    }
    gidx[(size_t)b * A_TOT + a] = (uint32_t)bg;
  }

  // per-gt top-10 by iterative min extraction (lowest-index tie break == top_k)
  const int wv = t >> 5, ln = t & 31;
  for (int g = wv; g < GTN; g += 8) {
    int sel[TOPKN];
    float cx = cgx[g], cy = cgy[g];
    for (int k = 0; k < TOPKN; ++k) {
      float best = 3.4e38f; int bi = 0x7fffffff;
      for (int a = ln; a < A_TOT; a += 32) {
        bool used = false;
        for (int j = 0; j < k; ++j) used |= (sel[j] == a);
        if (used) continue;
        int off, W, HW; float st;
        anchor_decode(a, &off, &W, &HW, &st);
        float ax = ((off % W) + 0.5f) * st, ay = ((off / W) + 0.5f) * st;
        float ddx = ax - cx, ddy = ay - cy;
        float d = ddx * ddx + ddy * ddy;
        if (d < best) { best = d; bi = a; }   // ascending a => lowest idx kept
      }
      for (int o = 16; o; o >>= 1) {          // butterfly: all lanes get global min
        float ov = __shfl_xor(best, o, 32);
        int   oi = __shfl_xor(bi, o, 32);
        if (ov < best || (ov == best && oi < bi)) { best = ov; bi = oi; }
      }
      sel[k] = bi;
      if (ln == 0) {
        atomicOr(&maskw[(size_t)b * MASK_W + (bi >> 5)], 1u << (bi & 31));
        int cls = gcls[b * GTN + g];
        int key = bi * NCLS + cls;
        atomicOr(&pairw[(size_t)b * PAIR_W + (key >> 5)], 1u << (key & 31));
      }
    }
  }
}

// ---------------------------------------------------------------- kernel 2: main pass
// grid = (B, 9), block 256 = 8 waves; each wave owns one 16-anchor tile.
// DFL staging uses gfx1250 async global->LDS copies (ASYNCcnt); DFL softmax
// reductions (sum, weighted sum) use v_wmma_f32_16x16x32_f16.
__global__ void k_main(const float* __restrict__ p3, const float* __restrict__ p4,
                       const float* __restrict__ p5, const float* __restrict__ gt,
                       const uint32_t* __restrict__ gidx, const uint32_t* __restrict__ maskw,
                       const uint32_t* __restrict__ pairw, float* __restrict__ accum) {
  __shared__ float    regL[8][16][66];   // raw DFL logits per tile
  __shared__ _Float16 exL[8][16][64];    // exp(x - max), f16, WMMA A staging
  __shared__ float    sums[8][16][8];    // WMMA D: {den,num} x 4 sides

  const int b  = blockIdx.x;
  const int wv = threadIdx.x >> 5, ln = threadIdx.x & 31;
  const int tile   = blockIdx.y * 8 + wv;
  const bool active = (tile < TILES);
  const int tl = active ? tile : 0;
  const int m  = ln & 15;
  int aa = tl * 16 + m;
  const bool valid = active && (aa < A_TOT);
  const int a = valid ? aa : (A_TOT - 1);

  int off, Wd, HW; float st;
  anchor_decode(a, &off, &Wd, &HW, &st);
  const float* lp = (a < 784) ? p3 : (a < 980) ? p4 : p5;
  const float* base = lp + (size_t)b * NO_CH * HW + off;  // channel c at base[c*HW]
  const float ax = (off % Wd) + 0.5f, ay = (off / Wd) + 0.5f;

  // pull the class-logit stream toward the WGP while the DFL phase runs
  const float* cb = base + (size_t)64 * HW;
  __builtin_prefetch(cb, 0, 0);

  // ---- stage 64 DFL channels straight into LDS via async copies (no VGPR trip)
  for (int it = 0; it < 32; ++it) {
    int c = 2 * it + (ln >> 4);
    uint32_t ldsoff = (uint32_t)(uintptr_t)&regL[wv][m][c];  // addrspace(3) offset
    const float* src = base + (size_t)c * HW;
    asm volatile("global_load_async_to_lds_b32 %0, %1, off"
                 :: "v"(ldsoff), "v"(src)
                 : "memory");
  }
  asm volatile("s_wait_asynccnt 0x0" ::: "memory");
  __syncthreads();

  // ---- softmax numerators in f16 (each lane does 2 (anchor,side) rows)
  for (int rr = 0; rr < 2; ++rr) {
    int r = ln + 32 * rr;
    int mm = r >> 2, s = r & 3;
    float mx = -3.4e38f;
    for (int k = 0; k < 16; ++k) mx = fmaxf(mx, regL[wv][mm][s * 16 + k]);
    for (int k = 0; k < 16; ++k)
      exL[wv][mm][s * 16 + k] = (_Float16)expf(regL[wv][mm][s * 16 + k] - mx);
  }
  __syncthreads();

  // ---- B matrix (32x16 f16): col0=ones(K<16), col1=proj(K<16), col2/3 same for K>=16
  const int nn = ln & 15, kb = (ln >> 4) * 16, hb = (ln >> 4) * 8;
  HV16 Bf;
  for (int e = 0; e < 16; ++e) {
    int k = (e < 8) ? hb + e : 8 + hb + e;
    float v = 0.f;
    if      (nn == 0) v = (k < 16)  ? 1.f : 0.f;
    else if (nn == 1) v = (k < 16)  ? (float)k : 0.f;
    else if (nn == 2) v = (k >= 16) ? 1.f : 0.f;
    else if (nn == 3) v = (k >= 16) ? (float)(k - 16) : 0.f;
    Bf.e[e] = (_Float16)v;
  }
  // ---- A matrices: 16 anchors x 32 bins (two sides per WMMA)
  HV16 A0, A1;
  for (int e = 0; e < 8; ++e) {
    A0.e[e]     = exL[wv][m][hb + e];          // side0 bins
    A0.e[8 + e] = exL[wv][m][16 + hb + e];     // side1 bins
    A1.e[e]     = exL[wv][m][32 + hb + e];     // side2 bins
    A1.e[8 + e] = exL[wv][m][48 + hb + e];     // side3 bins
  }
  FV8 D0, D1; D0.v = (v8f){}; D1.v = (v8f){};
  D0.v = __builtin_amdgcn_wmma_f32_16x16x32_f16(false, A0.v, false, Bf.v, (short)0, D0.v, false, false);
  D1.v = __builtin_amdgcn_wmma_f32_16x16x32_f16(false, A1.v, false, Bf.v, (short)0, D1.v, false, false);
  if (nn < 4) {   // D[m][n]: lanes 0-15 -> rows 0-7, lanes 16-31 -> rows 8-15
    int mb = (ln < 16) ? 0 : 8;
    for (int j = 0; j < 8; ++j) {
      sums[wv][mb + j][nn]     = D0.e[j];
      sums[wv][mb + j][4 + nn] = D1.e[j];
    }
  }
  __syncthreads();

  // ---- box loss (lanes 0-15, one anchor each)
  float bn = 0.f, bd = 0.f;
  if (ln < 16) {
    float dlx = sums[wv][m][1] / sums[wv][m][0];
    float dly = sums[wv][m][3] / sums[wv][m][2];
    float drx = sums[wv][m][5] / sums[wv][m][4];
    float dry = sums[wv][m][7] / sums[wv][m][6];
    float cxp = (ax + (drx - dlx) * 0.5f) * st;
    float cyp = (ay + (dry - dly) * 0.5f) * st;
    float pw  = (dlx + drx) * st;
    float ph  = (dly + dry) * st;
    uint32_t gi = gidx[(size_t)b * A_TOT + a];
    const float* gp = gt + ((size_t)b * GTN + gi) * 4;
    float iou = ciou_f(cxp, cyp, pw, ph,
                       gp[0] * 224.f, gp[1] * 224.f, gp[2] * 224.f, gp[3] * 224.f);
    float mf = (float)((maskw[(size_t)b * MASK_W + (a >> 5)] >> (a & 31)) & 1u);
    bn = (1.f - iou) * mf;
    bd = mf;
    if (!valid) { bn = 0.f; bd = 0.f; }
  }

  // ---- class BCE: softplus(x) - x*t with t from pair bitset (coalesced 16-wide)
  float ac = 0.f;
  for (int i = 0; i < 40; ++i) {
    int c = (ln >> 4) + 2 * i;
    float x = cb[(size_t)c * HW];
    int key = a * NCLS + c;
    float tt = (float)((pairw[(size_t)b * PAIR_W + (key >> 5)] >> (key & 31)) & 1u);
    ac += fmaxf(x, 0.f) - x * tt + log1pf(expf(-fabsf(x)));
  }
  if (!valid) ac = 0.f;

  ac = wave_sum(ac); bn = wave_sum(bn); bd = wave_sum(bd);
  if (ln == 0) {
    atomicAdd(&accum[b * 4 + 0], ac);
    atomicAdd(&accum[b * 4 + 1], bn);
    atomicAdd(&accum[b * 4 + 2], bd);
  }
}

// ---------------------------------------------------------------- kernel 3: final
__global__ void k_final(const float* __restrict__ accum, float* __restrict__ out, int B) {
  __shared__ float red[256];
  float s = 0.f;
  for (int b = threadIdx.x; b < B; b += 256) {
    float sc = accum[b * 4 + 0] * (0.5f / ((float)A_TOT * (float)NCLS));
    float sb = 7.5f * accum[b * 4 + 1] / fmaxf(accum[b * 4 + 2], 1.f);
    s += sc + sb;
  }
  red[threadIdx.x] = s;
  __syncthreads();
  for (int o = 128; o; o >>= 1) {
    if (threadIdx.x < o) red[threadIdx.x] += red[threadIdx.x + o];
    __syncthreads();
  }
  if (threadIdx.x == 0) out[0] = red[0] / (float)B;
}

// ---------------------------------------------------------------- launch
extern "C" void kernel_launch(void* const* d_in, const int* in_sizes, int n_in,
                              void* d_out, int out_size, void* d_ws, size_t ws_size,
                              hipStream_t stream) {
  const float* p3   = (const float*)d_in[0];
  const float* p4   = (const float*)d_in[1];
  const float* p5   = (const float*)d_in[2];
  const float* gt   = (const float*)d_in[3];
  const int*   gcls = (const int*)d_in[4];
  const int B = in_sizes[3] / (GTN * 4);

  // ws layout (32-bit words): accum | gidx | mask bits | pair bits  (~3.7 MB for B=256)
  uint32_t* wsw = (uint32_t*)d_ws;
  size_t accOff  = 0;
  size_t gidxOff = accOff  + (size_t)B * 4;
  size_t maskOff = gidxOff + (size_t)B * A_TOT;
  size_t pairOff = maskOff + (size_t)B * MASK_W;
  size_t total   = pairOff + (size_t)B * PAIR_W;
  float*    accum = (float*)(wsw + accOff);
  uint32_t* gidx  = wsw + gidxOff;
  uint32_t* maskw = wsw + maskOff;
  uint32_t* pairw = wsw + pairOff;

  k_init<<<1024, 256, 0, stream>>>(wsw, total);
  k_assign<<<B, 256, 0, stream>>>(gt, gcls, gidx, maskw, pairw);
  dim3 g2(B, (TILES + 7) / 8);
  k_main<<<g2, 256, 0, stream>>>(p3, p4, p5, gt, gidx, maskw, pairw, accum);
  k_final<<<1, 256, 0, stream>>>(accum, (float*)d_out, B);
}
